// LoretzFusion_86337432584665
// MI455X (gfx1250) — matrix operand order repaired and checked by reading, not compiled, not run on hardware
//
#include <hip/hip_runtime.h>
#include <hip/hip_bf16.h>
#include <cstdint>
#include <cstddef>

// ---------------------------------------------------------------------------
// Sizes from the reference
// ---------------------------------------------------------------------------
#define NN    4096
#define MMDIM 768
#define RR    256
#define CC    5

typedef _Float16 half_t;
typedef __attribute__((ext_vector_type(16))) _Float16 v16h;
typedef __attribute__((ext_vector_type(8)))  float    v8f;
typedef __attribute__((ext_vector_type(4)))  float    v4f;

// ---------------------------------------------------------------------------
// WMMA fragment loaders for D = A * B^T with both A and B stored row-major,
// K contiguous (wave32, v_wmma_f32_16x16x32_f16).
//
// A (16x32, f16): lane l<16 holds row M=l, halves [0..15]=K k0+0..15;
//                 lanes>=16: row l-16, K k0+16..31.
// B (32x16, f16): lane l<16 holds col N=l, halves [0..15]=K k0+0..15;
//                 lanes>=16: col l-16, K k0+16..31.
// Caller passes p already offset by the per-lane K shift.
// ---------------------------------------------------------------------------
__device__ __forceinline__ v16h ld_frag_a(const half_t* p) {
  v4f x0 = *(const v4f*)(p);         // 8 halves: K j..j+7
  v4f x1 = *(const v4f*)(p + 8);     // 8 halves: K j+8..j+15
  v16h r;
  __builtin_memcpy(&r, &x0, 16);
  __builtin_memcpy((char*)&r + 16, &x1, 16);
  return r;
}
__device__ __forceinline__ v16h ld_frag_b(const half_t* p) {
  v4f x0 = *(const v4f*)(p);         // 8 halves: K j..j+7
  v4f x1 = *(const v4f*)(p + 8);     // 8 halves: K j+8..j+15
  v16h r;
  __builtin_memcpy(&r, &x0, 16);
  __builtin_memcpy((char*)&r + 16, &x1, 16);
  return r;
}
#define WMMA_F16(a, b, c) \
  __builtin_amdgcn_wmma_f32_16x16x32_f16(false, (a), false, (b), (short)0, (c), false, false)

// ---------------------------------------------------------------------------
// CDNA5 async global->LDS copy (ASYNCcnt path, overlaps with VALU/WMMA).
// lds_off: wave-relative LDS byte address (low 32 bits of generic pointer).
// ---------------------------------------------------------------------------
__device__ __forceinline__ void async_copy_b128(unsigned lds_off, unsigned long long gaddr) {
  asm volatile("global_load_async_to_lds_b128 %0, %1, off"
               :: "v"(lds_off), "v"(gaddr) : "memory");
}
__device__ __forceinline__ void wait_async0() {
  asm volatile("s_wait_asynccnt 0x0" ::: "memory");
}
__device__ __forceinline__ unsigned lds_addr32(const void* p) {
  return (unsigned)(uintptr_t)p;  // LDS aperture: offset lives in addr[31:0]
}

// ---------------------------------------------------------------------------
// Utility kernels
// ---------------------------------------------------------------------------
__global__ void k_zero(float* __restrict__ p, int n) {
  int i = blockIdx.x * blockDim.x + threadIdx.x;
  if (i < n) p[i] = 0.0f;
}

__global__ void k_f2h(const float* __restrict__ s, half_t* __restrict__ d, int n) {
  int i = blockIdx.x * blockDim.x + threadIdx.x;
  if (i < n) d[i] = (half_t)s[i];
}

// ---------------------------------------------------------------------------
// Encoder GEMM1: h1[c,n,e] = gelu( sum_d mm[n,d] * W1[c,e,d] + b1[c,e] )
// ---------------------------------------------------------------------------
__global__ void __launch_bounds__(32)
k_gemm1(const half_t* __restrict__ Ah, const half_t* __restrict__ W1h,
        const float* __restrict__ b1, half_t* __restrict__ h1) {
  const int n0 = blockIdx.x * 16, e0 = blockIdx.y * 16, c = blockIdx.z;
  const int tid = threadIdx.x, ln = tid & 15, hi = tid >> 4;
  const half_t* B = W1h + (size_t)c * RR * MMDIM;
  const half_t* Arow = Ah + (size_t)(n0 + ln) * MMDIM + hi * 16;
  const half_t* Brow = B + (size_t)(e0 + ln) * MMDIM + hi * 16;
  v8f acc = {0.f, 0.f, 0.f, 0.f, 0.f, 0.f, 0.f, 0.f};
  for (int k0 = 0; k0 < MMDIM; k0 += 32) {
    v16h a = ld_frag_a(Arow + k0);
    v16h b = ld_frag_b(Brow + k0);
    acc = WMMA_F16(a, b, acc);
  }
  const int col = e0 + ln;
  const float bias = b1[c * RR + col];
#pragma unroll
  for (int v = 0; v < 8; v++) {
    const int row = n0 + v + 8 * hi;
    float x = acc[v] + bias;
    float g = 0.5f * x * (1.0f + erff(x * 0.70710678118654752f));  // exact GELU
    h1[((size_t)c * NN + row) * RR + col] = (half_t)g;
  }
}

// ---------------------------------------------------------------------------
// Encoder GEMM2: h2[c,n,f] = sum_e h1[c,n,e] * W2[c,f,e] + b2[c,f]  (f32 out)
// ---------------------------------------------------------------------------
__global__ void __launch_bounds__(32)
k_gemm2(const half_t* __restrict__ h1, const half_t* __restrict__ W2h,
        const float* __restrict__ b2, float* __restrict__ h2) {
  const int n0 = blockIdx.x * 16, f0 = blockIdx.y * 16, c = blockIdx.z;
  const int tid = threadIdx.x, ln = tid & 15, hi = tid >> 4;
  const half_t* A = h1 + (size_t)c * NN * RR;
  const half_t* B = W2h + (size_t)c * RR * RR;
  const half_t* Arow = A + (size_t)(n0 + ln) * RR + hi * 16;
  const half_t* Brow = B + (size_t)(f0 + ln) * RR + hi * 16;
  v8f acc = {0.f, 0.f, 0.f, 0.f, 0.f, 0.f, 0.f, 0.f};
  for (int k0 = 0; k0 < RR; k0 += 32) {
    v16h a = ld_frag_a(Arow + k0);
    v16h b = ld_frag_b(Brow + k0);
    acc = WMMA_F16(a, b, acc);
  }
  const int col = f0 + ln;
  const float bias = b2[c * RR + col];
#pragma unroll
  for (int v = 0; v < 8; v++) {
    const int row = n0 + v + 8 * hi;
    h2[((size_t)c * NN + row) * RR + col] = acc[v] + bias;
  }
}

// ---------------------------------------------------------------------------
// BatchNorm statistics (training-mode, biased var): partial sums via atomics.
// ---------------------------------------------------------------------------
__global__ void __launch_bounds__(256)
k_bnstats(const float* __restrict__ h2, float* __restrict__ bnacc) {
  const int c = blockIdx.y, chunk = blockIdx.x, f = threadIdx.x;
  const int rows = NN / 32;
  const float* base = h2 + ((size_t)c * NN + (size_t)chunk * rows) * RR + f;
  float s = 0.f, s2 = 0.f;
  for (int r = 0; r < rows; r++) {
    float x = base[(size_t)r * RR];
    s += x;
    s2 += x * x;
  }
  atomicAdd(&bnacc[c * RR + f], s);
  atomicAdd(&bnacc[CC * RR + c * RR + f], s2);
}

__global__ void __launch_bounds__(256)
k_bnfinal(const float* __restrict__ bnacc, float* __restrict__ bnmr) {
  const int idx = blockIdx.x * RR + threadIdx.x;  // c*R + f
  const float invN = 1.0f / (float)NN;
  float mu = bnacc[idx] * invN;
  float var = bnacc[CC * RR + idx] * invN - mu * mu;
  bnmr[idx] = mu;
  bnmr[CC * RR + idx] = rsqrtf(var + 1e-5f);
}

// ---------------------------------------------------------------------------
// BN apply + write mm_out (d_out, [N,R,C]) + row L2-normalize -> mmn_h (f16)
// ---------------------------------------------------------------------------
__global__ void __launch_bounds__(256)
k_bnapply(const float* __restrict__ h2, const float* __restrict__ bnmr,
          const float* __restrict__ gamma, const float* __restrict__ beta,
          float* __restrict__ out, half_t* __restrict__ mmn) {
  const int n = blockIdx.x, c = blockIdx.y, f = threadIdx.x;
  const size_t idx = ((size_t)c * NN + n) * RR + f;
  const int cf = c * RR + f;
  float y = (h2[idx] - bnmr[cf]) * bnmr[CC * RR + cf] * gamma[cf] + beta[cf];
  out[(size_t)n * (RR * CC) + f * CC + c] = y;
  __shared__ float red[RR];
  red[f] = y * y;
  __syncthreads();
  for (int s = RR / 2; s > 0; s >>= 1) {
    if (f < s) red[f] += red[f + s];
    __syncthreads();
  }
  float nrm = sqrtf(red[0]) + 1e-12f;
  mmn[idx] = (half_t)(y / nrm);
}

// ---------------------------------------------------------------------------
// st channel extract + row L2-normalize -> stn_h (f16)
// ---------------------------------------------------------------------------
__global__ void __launch_bounds__(256)
k_stnorm(const float* __restrict__ st, half_t* __restrict__ stn) {
  const int n = blockIdx.x, c = blockIdx.y, f = threadIdx.x;
  float x = st[(size_t)n * (RR * CC) + f * CC + c];
  __shared__ float red[RR];
  red[f] = x * x;
  __syncthreads();
  for (int s = RR / 2; s > 0; s >>= 1) {
    if (f < s) red[f] += red[f + s];
    __syncthreads();
  }
  float nrm = sqrtf(red[0]) + 1e-12f;
  stn[((size_t)c * NN + n) * RR + f] = (half_t)(x / nrm);
}

// ---------------------------------------------------------------------------
// Pass 1 over sm: fused rowsum( exp(2*cos) ) and row argmax(cos).
// Block = 4 waves, 64 output rows. A tiles (32KB) staged once in LDS.
// B tiles (8KB each, contiguous 16 rows of mmn) double-buffered in LDS via
// async global->LDS (ASYNCcnt), shared by all 4 waves -> 4x less L2 traffic,
// fill of tile t+1 overlaps WMMAs of tile t.
// ---------------------------------------------------------------------------
__global__ void __launch_bounds__(128)
k_simsum(const half_t* __restrict__ stn, const half_t* __restrict__ mmn,
         float* __restrict__ rowsum, int* __restrict__ amax) {
  __shared__ __align__(16) half_t At[64 * RR];       // 32 KB
  __shared__ __align__(16) half_t Bt[2][16 * RR];    // 2 x 8 KB
  const int c = blockIdx.y, n0 = blockIdx.x * 64;
  const int tid = threadIdx.x, w = tid >> 5, lane = tid & 31;
  const int ln = lane & 15, hi = lane >> 4;

  const half_t* S = stn + ((size_t)c * NN + n0) * RR;  // 64 contiguous rows
  for (int i = tid; i < 64 * RR / 8; i += 128)
    ((v4f*)At)[i] = ((const v4f*)S)[i];

  const half_t* M = mmn + (size_t)c * NN * RR;
  // prefetch tile 0 (rows 0..15 of mmn channel: one contiguous 8KB block)
  {
    const unsigned dst = lds_addr32(&Bt[0][0]);
    const unsigned long long src = (unsigned long long)(uintptr_t)M;
#pragma unroll
    for (int j = 0; j < 4; j++) {
      const int chunk = tid + j * 128;
      async_copy_b128(dst + chunk * 16, src + (unsigned long long)chunk * 16);
    }
  }
  wait_async0();
  __syncthreads();

  const half_t* Abase = At + (size_t)(w * 16 + ln) * RR + hi * 16;
  float rs[8], bv[8];
  int bi[8];
#pragma unroll
  for (int v = 0; v < 8; v++) { rs[v] = 0.f; bv[v] = -3.0e38f; bi[v] = 0; }

  const int T = NN / 16;
  for (int t = 0; t < T; t++) {
    if (t + 1 < T) {  // overlap: fill next buffer while computing this one
      const unsigned dst = lds_addr32(&Bt[(t + 1) & 1][0]);
      const unsigned long long src =
          (unsigned long long)(uintptr_t)(M + (size_t)(t + 1) * 16 * RR);
#pragma unroll
      for (int j = 0; j < 4; j++) {
        const int chunk = tid + j * 128;
        async_copy_b128(dst + chunk * 16, src + (unsigned long long)chunk * 16);
      }
    }
    const half_t* Bbase = Bt[t & 1] + (size_t)ln * RR + hi * 16;
    v8f acc = {0.f, 0.f, 0.f, 0.f, 0.f, 0.f, 0.f, 0.f};
#pragma unroll
    for (int kk = 0; kk < RR / 32; kk++) {
      v16h a = ld_frag_a(Abase + kk * 32);
      v16h b = ld_frag_b(Bbase + kk * 32);
      acc = WMMA_F16(a, b, acc);
    }
    const int mi = t * 16 + ln;
#pragma unroll
    for (int v = 0; v < 8; v++) {
      float cs = acc[v];
      rs[v] += __expf(2.0f * cs);  // tau = 0.5
      if (cs > bv[v]) { bv[v] = cs; bi[v] = mi; }
    }
    wait_async0();   // next tile landed
    __syncthreads(); // everyone done reading Bt[t&1]
  }

#pragma unroll
  for (int v = 0; v < 8; v++) {
    float s = rs[v];
    for (int m = 8; m >= 1; m >>= 1) s += __shfl_xor(s, m, 16);
    float xv = bv[v];
    int xi = bi[v];
    for (int m = 8; m >= 1; m >>= 1) {
      float ov = __shfl_xor(xv, m, 16);
      int oi = __shfl_xor(xi, m, 16);
      if (ov > xv || (ov == xv && oi < xi)) { xv = ov; xi = oi; }
    }
    if (ln == 0) {
      const int row = n0 + w * 16 + v + 8 * hi;
      rowsum[c * NN + row] = s;
      amax[c * NN + row] = xi;
    }
  }
}

// ---------------------------------------------------------------------------
// Pass 2: gathered rows q = mmn[argmax[n]]; stream q . mmn^T keeping per-row
// top-6 (val,idx) in registers, LDS merge, pos = sum exp(2*stn[n].mmn[idx]),
// loss term -log(pos/rowsum[n]). Same 4-wave async double-buffered B stream.
// ---------------------------------------------------------------------------
__global__ void __launch_bounds__(128)
k_topk(const half_t* __restrict__ stn, const half_t* __restrict__ mmn,
       const float* __restrict__ rowsum, const int* __restrict__ amax,
       float* __restrict__ loss) {
  __shared__ __align__(16) half_t At[64 * RR];     // 32 KB
  __shared__ __align__(16) half_t Bt[2][16 * RR];  // 16 KB
  __shared__ float cv[64 * 96];                    // 24 KB
  __shared__ int ci[64 * 96];                      // 24 KB
  __shared__ float tred[128];
  const int c = blockIdx.y, n0 = blockIdx.x * 64;
  const int tid = threadIdx.x, w = tid >> 5, lane = tid & 31;
  const int ln = lane & 15, hi = lane >> 4;
  const half_t* M = mmn + (size_t)c * NN * RR;

  // gather A rows: block row r -> mmn[amax[n0+r]]  (32 x 16B chunks per row)
  for (int i = tid; i < 64 * RR / 8; i += 128) {
    const int row = i >> 5;
    const int p = amax[c * NN + n0 + row];
    ((v4f*)At)[i] = ((const v4f*)(M + (size_t)p * RR))[i & 31];
  }
  {
    const unsigned dst = lds_addr32(&Bt[0][0]);
    const unsigned long long src = (unsigned long long)(uintptr_t)M;
#pragma unroll
    for (int j = 0; j < 4; j++) {
      const int chunk = tid + j * 128;
      async_copy_b128(dst + chunk * 16, src + (unsigned long long)chunk * 16);
    }
  }
  wait_async0();
  __syncthreads();

  const half_t* Abase = At + (size_t)(w * 16 + ln) * RR + hi * 16;
  float tv[48];
  int ti[48];
#pragma unroll
  for (int i = 0; i < 48; i++) { tv[i] = -3.0e38f; ti[i] = 0; }

  const int T = NN / 16;
  for (int t = 0; t < T; t++) {
    if (t + 1 < T) {
      const unsigned dst = lds_addr32(&Bt[(t + 1) & 1][0]);
      const unsigned long long src =
          (unsigned long long)(uintptr_t)(M + (size_t)(t + 1) * 16 * RR);
#pragma unroll
      for (int j = 0; j < 4; j++) {
        const int chunk = tid + j * 128;
        async_copy_b128(dst + chunk * 16, src + (unsigned long long)chunk * 16);
      }
    }
    const half_t* Bbase = Bt[t & 1] + (size_t)ln * RR + hi * 16;
    v8f acc = {0.f, 0.f, 0.f, 0.f, 0.f, 0.f, 0.f, 0.f};
#pragma unroll
    for (int kk = 0; kk < RR / 32; kk++) {
      v16h a = ld_frag_a(Abase + kk * 32);
      v16h b = ld_frag_b(Bbase + kk * 32);
      acc = WMMA_F16(a, b, acc);
    }
#pragma unroll
    for (int v = 0; v < 8; v++) {
      float x = acc[v];
      int xi = t * 16 + ln;
#pragma unroll
      for (int j = 0; j < 6; j++) {  // sorted-descending insertion
        const int s = v * 6 + j;
        if (x > tv[s]) {
          float tf = tv[s]; tv[s] = x; x = tf;
          int t2 = ti[s]; ti[s] = xi; xi = t2;
        }
      }
    }
    wait_async0();
    __syncthreads();
  }

  // dump per-lane top-6 candidates: 16 lanes * 6 = 96 per row
#pragma unroll
  for (int v = 0; v < 8; v++) {
    const int row = w * 16 + v + 8 * hi;
    const int base = row * 96 + ln * 6;
#pragma unroll
    for (int j = 0; j < 6; j++) {
      cv[base + j] = tv[v * 6 + j];
      ci[base + j] = ti[v * 6 + j];
    }
  }
  __syncthreads();

  float term = 0.f;
  if (tid < 64) {
    const int n = n0 + tid;
    float bv6[6];
    int bi6[6];
#pragma unroll
    for (int j = 0; j < 6; j++) { bv6[j] = -3.0e38f; bi6[j] = 0; }
    for (int k = 0; k < 96; k++) {
      float x = cv[tid * 96 + k];
      int xi = ci[tid * 96 + k];
#pragma unroll
      for (int j = 0; j < 6; j++) {
        if (x > bv6[j]) {
          float tf = bv6[j]; bv6[j] = x; x = tf;
          int t2 = bi6[j]; bi6[j] = xi; xi = t2;
        }
      }
    }
    const half_t* sp = stn + ((size_t)c * NN + n) * RR;
    float pos = 0.f;
    for (int j = 0; j < 6; j++) {
      const half_t* mp = M + (size_t)bi6[j] * RR;
      float d = 0.f;
      for (int k = 0; k < RR; k++) d += (float)sp[k] * (float)mp[k];
      pos += __expf(2.0f * d);
    }
    term = __logf(rowsum[c * NN + n]) - __logf(pos);  // -log(pos/sum)
  }
  tred[tid] = term;
  __syncthreads();
  for (int s = 64; s > 0; s >>= 1) {
    if (tid < s) tred[tid] += tred[tid + s];
    __syncthreads();
  }
  if (tid == 0) atomicAdd(loss, tred[0] * (1.0f / ((float)NN * (float)CC)));
}

// ---------------------------------------------------------------------------
// Launch
// ---------------------------------------------------------------------------
extern "C" void kernel_launch(void* const* d_in, const int* in_sizes, int n_in,
                              void* d_out, int out_size, void* d_ws, size_t ws_size,
                              hipStream_t stream) {
  const float* st = (const float*)d_in[0];
  const float* mm = (const float*)d_in[1];
  const float* W1 = (const float*)d_in[2];
  const float* b1 = (const float*)d_in[3];
  const float* W2 = (const float*)d_in[4];
  const float* b2 = (const float*)d_in[5];
  const float* gamma = (const float*)d_in[6];
  const float* beta = (const float*)d_in[7];
  float* out = (float*)d_out;
  float* loss = out + (out_size - 1);

  char* w = (char*)d_ws;
  auto alloc = [&](size_t bytes) -> void* {
    void* p = (void*)w;
    w += (bytes + 255) & ~(size_t)255;
    return p;
  };
  half_t* mm_h  = (half_t*)alloc((size_t)NN * MMDIM * 2);
  half_t* W1_h  = (half_t*)alloc((size_t)CC * RR * MMDIM * 2);
  half_t* W2_h  = (half_t*)alloc((size_t)CC * RR * RR * 2);
  half_t* h1_h  = (half_t*)alloc((size_t)CC * NN * RR * 2);
  float*  h2_f  = (float*)alloc((size_t)CC * NN * RR * 4);
  half_t* stn_h = (half_t*)alloc((size_t)CC * NN * RR * 2);
  half_t* mmn_h = (half_t*)alloc((size_t)CC * NN * RR * 2);
  float*  bnacc = (float*)alloc((size_t)2 * CC * RR * 4);
  float*  bnmr  = (float*)alloc((size_t)2 * CC * RR * 4);
  float*  rowsum = (float*)alloc((size_t)CC * NN * 4);
  int*    amax  = (int*)alloc((size_t)CC * NN * 4);

  // zero accumulators (harness poisons buffers; must re-zero every call)
  k_zero<<<dim3((2 * CC * RR + 255) / 256), 256, 0, stream>>>(bnacc, 2 * CC * RR);
  k_zero<<<1, 32, 0, stream>>>(loss, 1);

  // f32 -> f16 conversions
  const int nmm = NN * MMDIM, nw1 = CC * RR * MMDIM, nw2 = CC * RR * RR;
  k_f2h<<<dim3((nmm + 255) / 256), 256, 0, stream>>>(mm, mm_h, nmm);
  k_f2h<<<dim3((nw1 + 255) / 256), 256, 0, stream>>>(W1, W1_h, nw1);
  k_f2h<<<dim3((nw2 + 255) / 256), 256, 0, stream>>>(W2, W2_h, nw2);

  // encoders
  k_gemm1<<<dim3(NN / 16, RR / 16, CC), 32, 0, stream>>>(mm_h, W1_h, b1, h1_h);
  k_gemm2<<<dim3(NN / 16, RR / 16, CC), 32, 0, stream>>>(h1_h, W2_h, b2, h2_f);
  k_bnstats<<<dim3(32, CC), 256, 0, stream>>>(h2_f, bnacc);
  k_bnfinal<<<CC, RR, 0, stream>>>(bnacc, bnmr);
  k_bnapply<<<dim3(NN, CC), RR, 0, stream>>>(h2_f, bnmr, gamma, beta, out, mmn_h);
  k_stnorm<<<dim3(NN, CC), RR, 0, stream>>>(st, stn_h);

  // loss (4 waves/block, 64 rows/block, async double-buffered B stream)
  k_simsum<<<dim3(NN / 64, CC), 128, 0, stream>>>(stn_h, mmn_h, rowsum, amax);
  k_topk<<<dim3(NN / 64, CC), 128, 0, stream>>>(stn_h, mmn_h, rowsum, amax, loss);
}
